// PartSegmentationEmbHead_74105365725717
// MI455X (gfx1250) — matrix operand order repaired and checked by reading, not compiled, not run on hardware
//
#include <hip/hip_runtime.h>

typedef __attribute__((ext_vector_type(16))) __bf16 v16bf;
typedef __attribute__((ext_vector_type(8)))  __bf16 v8bf;
typedef __attribute__((ext_vector_type(4)))  __bf16 v4bf;
typedef __attribute__((ext_vector_type(8)))  float  v8f;
typedef __attribute__((ext_vector_type(4)))  unsigned int u32x4;
typedef __attribute__((ext_vector_type(8)))  int i32x8;
typedef __attribute__((ext_vector_type(4)))  int i32x4;

#define BB   4
#define NN   8192
#define SS   512
#define EE   512
#define CIN  1536
#define MM   512
#define ROWS (BB * NN)   // 32768

// ---------------------------------------------------------------------------
// fp32 -> bf16 (vectorized x4) for weight conversion
// ---------------------------------------------------------------------------
__global__ __launch_bounds__(256) void f32x4_to_bf16(const float* __restrict__ in,
                                                     __bf16* __restrict__ out, int n4) {
    int i = blockIdx.x * 256 + threadIdx.x;
    if (i < n4) {
        float4 x = ((const float4*)in)[i];
        v4bf y;
        y[0] = (__bf16)x.x; y[1] = (__bf16)x.y; y[2] = (__bf16)x.z; y[3] = (__bf16)x.w;
        ((v4bf*)out)[i] = y;
    }
}

// ---------------------------------------------------------------------------
// 3-NN search: one thread per point, centers cached in LDS (6 KB)
// ---------------------------------------------------------------------------
__global__ __launch_bounds__(256) void knn_kernel(const float* __restrict__ xyz,
                                                  const float* __restrict__ centers,
                                                  int* __restrict__ knn_idx,
                                                  float* __restrict__ knn_w) {
    __shared__ float cs[SS * 3];
    const int b = blockIdx.x >> 5;        // NN/256 = 32 blocks per batch
    const int chunk = blockIdx.x & 31;
    const int t = threadIdx.x;
    for (int i = t; i < SS * 3; i += 256) cs[i] = centers[b * SS * 3 + i];
    __syncthreads();
    const int p = b * NN + chunk * 256 + t;
    const float px = xyz[p * 3 + 0], py = xyz[p * 3 + 1], pz = xyz[p * 3 + 2];
    float d0 = 3.4e38f, d1 = 3.4e38f, d2v = 3.4e38f;
    int i0 = 0, i1 = 0, i2 = 0;
    for (int s = 0; s < SS; ++s) {
        float dx = px - cs[s * 3 + 0];
        float dy = py - cs[s * 3 + 1];
        float dz = pz - cs[s * 3 + 2];
        float d = dx * dx + dy * dy + dz * dz;
        if (d < d0)       { d2v = d1; i2 = i1; d1 = d0; i1 = i0; d0 = d; i0 = s; }
        else if (d < d1)  { d2v = d1; i2 = i1; d1 = d; i1 = s; }
        else if (d < d2v) { d2v = d; i2 = s; }
    }
    float r0 = 1.0f / (d0 + 1e-8f), r1 = 1.0f / (d1 + 1e-8f), r2 = 1.0f / (d2v + 1e-8f);
    float rs = 1.0f / (r0 + r1 + r2);
    knn_idx[p * 3 + 0] = i0; knn_idx[p * 3 + 1] = i1; knn_idx[p * 3 + 2] = i2;
    knn_w[p * 3 + 0] = r0 * rs; knn_w[p * 3 + 1] = r1 * rs; knn_w[p * 3 + 2] = r2 * rs;
}

// ---------------------------------------------------------------------------
// Distance-weighted interpolation of concat(H4,H8,H12); emits bf16 GEMM input.
// One block per point; H* rows are L2-resident (12.6 MB total).
// ---------------------------------------------------------------------------
__global__ __launch_bounds__(256) void interp_kernel(const float* __restrict__ H4,
                                                     const float* __restrict__ H8,
                                                     const float* __restrict__ H12,
                                                     const int* __restrict__ knn_idx,
                                                     const float* __restrict__ knn_w,
                                                     __bf16* __restrict__ interp) {
    const int p = blockIdx.x;
    const int b = p >> 13;                 // / NN
    __shared__ int   si[3];
    __shared__ float sw[3];
    if (threadIdx.x < 3) {
        si[threadIdx.x] = knn_idx[p * 3 + threadIdx.x];
        sw[threadIdx.x] = knn_w[p * 3 + threadIdx.x];
    }
    __syncthreads();
    const int r0 = b * SS + si[0], r1 = b * SS + si[1], r2 = b * SS + si[2];
    const float w0 = sw[0], w1 = sw[1], w2 = sw[2];
    for (int c = threadIdx.x; c < CIN; c += 256) {
        const float* src = (c < EE) ? H4 : (c < 2 * EE) ? H8 : H12;
        const int cc = c & (EE - 1);
        float v = w0 * src[r0 * EE + cc] + w1 * src[r1 * EE + cc] + w2 * src[r2 * EE + cc];
        interp[(size_t)p * CIN + c] = (__bf16)v;
    }
}

// ---------------------------------------------------------------------------
// Tensor Data Mover: 2D bf16 tile (tile_d0 K-elems x tile_d1 rows) -> LDS.
// D# pad feature inserts 16 B after every 64 B tile row => 80 B LDS pitch
// (LDT = 40 bf16), matching the bank-conflict-avoiding WMMA fragment reads.
// ---------------------------------------------------------------------------
__device__ __forceinline__ unsigned lds_off_of(const void* p) {
    return (unsigned)(unsigned long long)p;   // flat LDS addr[31:0] == LDS offset
}

__device__ __forceinline__ void tdm_load_2d(unsigned lds_off, const __bf16* gptr,
                                            unsigned tensor_d0, unsigned tensor_d1,
                                            unsigned tile_d0, unsigned tile_d1,
                                            unsigned stride_elems) {
    unsigned long long ga = (unsigned long long)gptr;
    u32x4 g0;
    g0[0] = 1u;                                  // count=1 (valid user descriptor)
    g0[1] = lds_off;                             // lds_addr [63:32]
    g0[2] = (unsigned)ga;                        // global_addr [95:64]
    g0[3] = (unsigned)(ga >> 32) | (2u << 30);   // global_addr[56:32] | type=2
    i32x8 g1;
    g1[0] = (int)((1u << 16)                     // data_size = 2 bytes
                | (1u << 20)                     // pad_enable
                | (3u << 22)                     // pad_interval: 16 DWORDs (64 B)
                | (3u << 25));                   // pad_amount: 4 DWORDs (16 B)
    g1[1] = (int)((tensor_d0 & 0xFFFFu) << 16);
    g1[2] = (int)(((tensor_d0 >> 16) & 0xFFFFu) | ((tensor_d1 & 0xFFFFu) << 16));
    g1[3] = (int)(((tensor_d1 >> 16) & 0xFFFFu) | (tile_d0 << 16));
    g1[4] = (int)(tile_d1 & 0xFFFFu);            // tile_dim2 = 0 (2D tile)
    g1[5] = (int)stride_elems;                   // tensor_dim0_stride [31:0]
    g1[6] = 0;
    g1[7] = 0;
    i32x4 z4; z4[0] = 0; z4[1] = 0; z4[2] = 0; z4[3] = 0;
    i32x8 z8;
#pragma unroll
    for (int i = 0; i < 8; ++i) z8[i] = 0;
    // amdgpu-toolchain (clang-23) 6-arg form
    __builtin_amdgcn_tensor_load_to_lds(g0, g1, z4, z4, z8, 0);
}

// ---------------------------------------------------------------------------
// bf16 WMMA GEMM: C[ROWS, MM] = A[ROWS, K] * W[MM, K]^T, fp32 accumulate.
// Block tile 128x64, 8 waves each computing 32x32 (2x2 WMMA tiles).
// TDM double-buffered LDS staging (wave 0 issues DMA, TENSORcnt-synced).
// Fused per-channel sum / sum-of-squares for BatchNorm batch statistics.
// ---------------------------------------------------------------------------
template <int K>
__global__ __launch_bounds__(256) void gemm_bf16_bn(const __bf16* __restrict__ A,
                                                    const __bf16* __restrict__ W,
                                                    float* __restrict__ C,
                                                    float* __restrict__ chsum,
                                                    float* __restrict__ chsq) {
    constexpr int LDT = 40;                     // 80-byte LDS row pitch (via TDM pad)
    __shared__ __bf16 As[2][128 * LDT];         // 2 x 10.0 KB
    __shared__ __bf16 Bs[2][64 * LDT];          // 2 x  5.0 KB
    const int t = threadIdx.x;
    const int lane = t & 31, wid = t >> 5;
    const int wm = (wid & 3) << 5;              // wave M offset: 0,32,64,96
    const int wn = (wid >> 2) << 5;             // wave N offset: 0,32
    const int rowBase = blockIdx.x << 7;
    const int colBase = blockIdx.y << 6;
    const int half = lane >> 4, l15 = lane & 15;
    const int kselA = half << 3;                // A lanes: K {0-7,16-23} / {8-15,24-31}
    const int kselB = half << 3;                // B lanes: K {0-7,16-23} / {8-15,24-31}

    v8f acc[2][2];
#pragma unroll
    for (int mi = 0; mi < 2; ++mi)
#pragma unroll
        for (int ni = 0; ni < 2; ++ni)
#pragma unroll
            for (int r = 0; r < 8; ++r) acc[mi][ni][r] = 0.0f;

    const __bf16* Abase = A + (size_t)rowBase * K;
    const __bf16* Bbase = W + (size_t)colBase * K;

    // Prologue: stage K-tile 0 into buffer 0
    if (wid == 0) {
        tdm_load_2d(lds_off_of(&As[0][0]), Abase, K, ROWS, 32, 128, K);
        tdm_load_2d(lds_off_of(&Bs[0][0]), Bbase, K, MM, 32, 64, K);
    }

    for (int k0 = 0; k0 < K; k0 += 32) {
        const int cur = (k0 >> 5) & 1;
        if (wid == 0) {
            if (k0 + 32 < K) {   // prefetch next K-tile into the other buffer
                tdm_load_2d(lds_off_of(&As[cur ^ 1][0]), Abase + k0 + 32, K, ROWS, 32, 128, K);
                tdm_load_2d(lds_off_of(&Bs[cur ^ 1][0]), Bbase + k0 + 32, K, MM, 32, 64, K);
                __builtin_amdgcn_s_wait_tensorcnt((short)2);  // in-order: current tile done
            } else {
                __builtin_amdgcn_s_wait_tensorcnt((short)0);
            }
        }
        __syncthreads();   // current tile visible to all waves

        v16bf af[2], bf[2];
#pragma unroll
        for (int mi = 0; mi < 2; ++mi) {
            const __bf16* p = &As[cur][(wm + (mi << 4) + l15) * LDT + kselA];
            v8bf lo = *(const v8bf*)p;
            v8bf hi = *(const v8bf*)(p + 16);
#pragma unroll
            for (int j = 0; j < 8; ++j) { af[mi][j] = lo[j]; af[mi][j + 8] = hi[j]; }
        }
#pragma unroll
        for (int ni = 0; ni < 2; ++ni) {
            const __bf16* p = &Bs[cur][(wn + (ni << 4) + l15) * LDT + kselB];
            v8bf lo = *(const v8bf*)p;
            v8bf hi = *(const v8bf*)(p + 16);
#pragma unroll
            for (int j = 0; j < 8; ++j) { bf[ni][j] = lo[j]; bf[ni][j + 8] = hi[j]; }
        }
#pragma unroll
        for (int mi = 0; mi < 2; ++mi)
#pragma unroll
            for (int ni = 0; ni < 2; ++ni)
                acc[mi][ni] = __builtin_amdgcn_wmma_f32_16x16x32_bf16(
                    false, af[mi], false, bf[ni], (short)0, acc[mi][ni], false, false);

        __syncthreads();   // reads done before this buffer is DMA-overwritten
    }

    // Epilogue: store pre-BN fp32 + fused per-channel BN statistics
#pragma unroll
    for (int ni = 0; ni < 2; ++ni) {
        const int col = colBase + wn + (ni << 4) + l15;
        float s = 0.0f, s2 = 0.0f;
#pragma unroll
        for (int mi = 0; mi < 2; ++mi) {
            const int rb = rowBase + wm + (mi << 4) + (half << 3);
#pragma unroll
            for (int r = 0; r < 8; ++r) {
                float v = acc[mi][ni][r];
                C[(size_t)(rb + r) * MM + col] = v;
                s += v; s2 += v * v;
            }
        }
        s  += __shfl_xor(s, 16, 32);     // lanes l and l^16 share the same channel
        s2 += __shfl_xor(s2, 16, 32);
        if (half == 0) {
            atomicAdd(&chsum[col], s);
            atomicAdd(&chsq[col], s2);
        }
    }
}

// ---------------------------------------------------------------------------
// BN finalize: per-channel scale/shift from batch stats
// ---------------------------------------------------------------------------
__global__ __launch_bounds__(256) void bn_finalize_kernel(const float* __restrict__ chsum,
                                                          const float* __restrict__ chsq,
                                                          const float* __restrict__ gamma,
                                                          const float* __restrict__ beta,
                                                          float* __restrict__ scale,
                                                          float* __restrict__ shift) {
    int c = blockIdx.x * 256 + threadIdx.x;
    if (c < MM) {
        const float inv = 1.0f / (float)ROWS;
        float mean = chsum[c] * inv;
        float var  = chsq[c] * inv - mean * mean;
        float sc   = gamma[c] * rsqrtf(var + 1e-5f);
        scale[c] = sc;
        shift[c] = beta[c] - mean * sc;
    }
}

// ---------------------------------------------------------------------------
// BN + ReLU apply (layer 1): fp32 -> bf16 for the next GEMM
// ---------------------------------------------------------------------------
__global__ __launch_bounds__(256) void bn_relu_to_bf16(const float* __restrict__ X,
                                                       const float* __restrict__ scale,
                                                       const float* __restrict__ shift,
                                                       __bf16* __restrict__ Y) {
    size_t i = ((size_t)blockIdx.x * 256 + threadIdx.x) * 4;
    const float4 x  = *(const float4*)(X + i);
    const int c     = (int)(i & (MM - 1));
    const float4 sc = *(const float4*)(scale + c);
    const float4 sh = *(const float4*)(shift + c);
    v4bf y;
    y[0] = (__bf16)fmaxf(fmaf(x.x, sc.x, sh.x), 0.0f);
    y[1] = (__bf16)fmaxf(fmaf(x.y, sc.y, sh.y), 0.0f);
    y[2] = (__bf16)fmaxf(fmaf(x.z, sc.z, sh.z), 0.0f);
    y[3] = (__bf16)fmaxf(fmaf(x.w, sc.w, sh.w), 0.0f);
    *(v4bf*)(Y + i) = y;
}

// ---------------------------------------------------------------------------
// BN + ReLU apply (layer 2): in-place fp32 on d_out
// ---------------------------------------------------------------------------
__global__ __launch_bounds__(256) void bn_relu_inplace_f32(float* __restrict__ X,
                                                           const float* __restrict__ scale,
                                                           const float* __restrict__ shift) {
    size_t i = ((size_t)blockIdx.x * 256 + threadIdx.x) * 4;
    float4 x        = *(const float4*)(X + i);
    const int c     = (int)(i & (MM - 1));
    const float4 sc = *(const float4*)(scale + c);
    const float4 sh = *(const float4*)(shift + c);
    x.x = fmaxf(fmaf(x.x, sc.x, sh.x), 0.0f);
    x.y = fmaxf(fmaf(x.y, sc.y, sh.y), 0.0f);
    x.z = fmaxf(fmaf(x.z, sc.z, sh.z), 0.0f);
    x.w = fmaxf(fmaf(x.w, sc.w, sh.w), 0.0f);
    *(float4*)(X + i) = x;
}

// ---------------------------------------------------------------------------
// Host-side orchestration (graph-capture safe: only async ops on `stream`)
// ---------------------------------------------------------------------------
extern "C" void kernel_launch(void* const* d_in, const int* in_sizes, int n_in,
                              void* d_out, int out_size, void* d_ws, size_t ws_size,
                              hipStream_t stream) {
    (void)in_sizes; (void)n_in; (void)out_size; (void)ws_size;

    const float* xyz     = (const float*)d_in[0];
    const float* centers = (const float*)d_in[1];
    const float* H4      = (const float*)d_in[2];
    const float* H8      = (const float*)d_in[3];
    const float* H12     = (const float*)d_in[4];
    const float* W1      = (const float*)d_in[5];
    // d_in[6] (b1) and d_in[10] (b2) cancel exactly under batch-statistics BN.
    const float* g1      = (const float*)d_in[7];
    const float* beta1   = (const float*)d_in[8];
    const float* W2      = (const float*)d_in[9];
    const float* g2      = (const float*)d_in[11];
    const float* beta2   = (const float*)d_in[12];
    float* OUT = (float*)d_out;

    uint8_t* w = (uint8_t*)d_ws;
    auto take = [&](size_t bytes) -> uint8_t* {
        uint8_t* p = w;
        w += (bytes + 255) & ~(size_t)255;
        return p;
    };
    __bf16* W1b    = (__bf16*)take((size_t)MM * CIN * 2);
    __bf16* W2b    = (__bf16*)take((size_t)MM * MM * 2);
    int*    kidx   = (int*)   take((size_t)ROWS * 3 * 4);
    float*  kw     = (float*) take((size_t)ROWS * 3 * 4);
    __bf16* interp = (__bf16*)take((size_t)ROWS * CIN * 2);
    float*  X1     = (float*) take((size_t)ROWS * MM * 4);
    __bf16* A2     = (__bf16*)take((size_t)ROWS * MM * 2);
    float*  stats  = (float*) take((size_t)8 * MM * 4);
    float* sum1 = stats,          *sq1 = stats + MM;
    float* sum2 = stats + 2 * MM, *sq2 = stats + 3 * MM;
    float* sc1  = stats + 4 * MM, *sh1 = stats + 5 * MM;
    float* sc2  = stats + 6 * MM, *sh2 = stats + 7 * MM;

    (void)hipMemsetAsync(stats, 0, (size_t)4 * MM * sizeof(float), stream);

    f32x4_to_bf16<<<(MM * CIN / 4 + 255) / 256, 256, 0, stream>>>(W1, W1b, MM * CIN / 4);
    f32x4_to_bf16<<<(MM * MM / 4 + 255) / 256, 256, 0, stream>>>(W2, W2b, MM * MM / 4);

    knn_kernel<<<BB * (NN / 256), 256, 0, stream>>>(xyz, centers, kidx, kw);
    interp_kernel<<<ROWS, 256, 0, stream>>>(H4, H8, H12, kidx, kw, interp);

    gemm_bf16_bn<CIN><<<dim3(ROWS / 128, MM / 64), 256, 0, stream>>>(interp, W1b, X1, sum1, sq1);
    bn_finalize_kernel<<<2, 256, 0, stream>>>(sum1, sq1, g1, beta1, sc1, sh1);
    bn_relu_to_bf16<<<ROWS * MM / 1024, 256, 0, stream>>>(X1, sc1, sh1, A2);

    gemm_bf16_bn<MM><<<dim3(ROWS / 128, MM / 64), 256, 0, stream>>>(A2, W2b, OUT, sum2, sq2);
    bn_finalize_kernel<<<2, 256, 0, stream>>>(sum2, sq2, g2, beta2, sc2, sh2);
    bn_relu_inplace_f32<<<ROWS * MM / 1024, 256, 0, stream>>>(OUT, sc2, sh2);
}
